// Grok1DecoderLayer_44753559224970
// MI455X (gfx1250) — compile-verified, matrix operand order checked by reading, not run on hardware
//
#include <hip/hip_runtime.h>
#include <hip/hip_bf16.h>

typedef __bf16 bf16;
typedef bf16  v16bf __attribute__((ext_vector_type(16)));
typedef bf16  v8bf  __attribute__((ext_vector_type(8)));
typedef float v8f   __attribute__((ext_vector_type(8)));
typedef float v4f   __attribute__((ext_vector_type(4)));
typedef unsigned int v4u __attribute__((ext_vector_type(4)));
typedef int   v4i  __attribute__((ext_vector_type(4)));
typedef int   v8i  __attribute__((ext_vector_type(8)));

namespace {
constexpr int T_ = 2048, H_ = 2048, NH_ = 32, NKV_ = 8, HD_ = 64;
constexpr int E_ = 8, I_ = 4096;
constexpr int QKVN_ = (NH_ + 2 * NKV_) * HD_;   // 3072
constexpr float EPS_ = 1e-5f;
}

__device__ __forceinline__ v8f wmma_bf16(v16bf a, v16bf b, v8f c) {
  return __builtin_amdgcn_wmma_f32_16x16x32_bf16(
      false, a, false, b, (short)0, c, false, false);
}

// fp32 -> bf16 convert of 16 contiguous values (weights converted in-flight;
// the 36us of fp32 weight HBM traffic is far below the WMMA time, so no
// precast scratch is needed).
__device__ __forceinline__ void cvt16(const float* __restrict__ src,
                                      v8bf& lo, v8bf& hi) {
  v4f f0 = *(const v4f*)(src + 0);
  v4f f1 = *(const v4f*)(src + 4);
  v4f f2 = *(const v4f*)(src + 8);
  v4f f3 = *(const v4f*)(src + 12);
  lo[0] = (bf16)f0[0]; lo[1] = (bf16)f0[1]; lo[2] = (bf16)f0[2]; lo[3] = (bf16)f0[3];
  lo[4] = (bf16)f1[0]; lo[5] = (bf16)f1[1]; lo[6] = (bf16)f1[2]; lo[7] = (bf16)f1[3];
  hi[0] = (bf16)f2[0]; hi[1] = (bf16)f2[1]; hi[2] = (bf16)f2[2]; hi[3] = (bf16)f2[3];
  hi[4] = (bf16)f3[0]; hi[5] = (bf16)f3[1]; hi[6] = (bf16)f3[2]; hi[7] = (bf16)f3[3];
}

// Issue a TDM 2D tile load: rows x 32 bf16 elements, row stride ldb elements,
// into LDS at lds_addr. D# dword packing per CDNA5 ISA ch.8 (group0/group1).
__device__ __forceinline__ void tdm_load_tile_bf16(
    unsigned long long gaddr, unsigned lds_addr, unsigned rows,
    unsigned long long ldb_elems) {
  v4u g0;
  g0[0] = 1u;                                          // count=1, user mode
  g0[1] = lds_addr;                                    // [63:32] lds_addr
  g0[2] = (unsigned)(gaddr & 0xffffffffu);             // [95:64] global lo
  g0[3] = (unsigned)((gaddr >> 32) & 0x1ffffffu) | (2u << 30); // addr hi | type=2
  v8i g1;
  unsigned td0 = 32u;                                  // tensor_dim0 (elems)
  g1[0] = (int)(1u << 16);                             // data_size=1 (2B), mask=0
  g1[1] = (int)((td0 & 0xffffu) << 16);                // dim0[15:0]
  g1[2] = (int)(((td0 >> 16) & 0xffffu) | ((rows & 0xffffu) << 16)); // dim0 hi|dim1 lo
  g1[3] = (int)(((rows >> 16) & 0xffffu) | (32u << 16));             // dim1 hi|tile_dim0=32
  g1[4] = (int)128u;                                   // tile_dim1=128, tile_dim2=0
  g1[5] = (int)(ldb_elems & 0xffffffffu);              // dim0_stride lo
  g1[6] = (int)((ldb_elems >> 32) & 0xffffu);          // dim0_stride hi | dim1_stride lo
  g1[7] = 0;
  v4i z4 = {0, 0, 0, 0};
#if __clang_major__ >= 23
  v8i z8 = {0, 0, 0, 0, 0, 0, 0, 0};
  __builtin_amdgcn_tensor_load_to_lds(g0, g1, z4, z4, z8, 0);
#else
  __builtin_amdgcn_tensor_load_to_lds(g0, g1, z4, z4, 0);
#endif
}

// ---------------------------------------------------------------------------
// Generic TN GEMM: C[M,N] = alpha * A[M,K](bf16) x B[N,K]^T.
// 256 threads = 8 waves as 4(M) x 2(N); block tile 128x128; wave tile 32x64
// (2 A frags x 4 B frags -> 8 WMMA per K-step). B tile (128x32) is double-
// buffered in LDS: bf16 B comes in via the Tensor Data Mover (one issuing
// wave, s_wait_tensorcnt + one barrier per K-step); fp32 B (weights) is
// cooperatively loaded + converted with a register double buffer.
// Epilogue fuses causal mask (with whole-block skip above the diagonal),
// per-row scale (MoE routing weight) and accumulation.
// ---------------------------------------------------------------------------
template <bool B_F32>
__global__ __launch_bounds__(256) void wmma_gemm_tn(
    const bf16* __restrict__ A, int lda,
    const void* __restrict__ Bv, int ldb,
    float* __restrict__ C, int ldc,
    int M, int N, int Kd, float alpha, int causal,
    const float* __restrict__ row_scale, int rs_stride, int accum)
{
  __shared__ __align__(128) bf16 Bs[2][128 * 32];
  const int tid  = threadIdx.x;
  const int wave = tid >> 5;
  const int lane = tid & 31;
  const int bm   = blockIdx.x * 128;
  const int bn   = blockIdx.y * 128;
  const int m0   = bm + (wave >> 1) * 32;
  const int wn   = wave & 1;
  const int n0   = bn + wn * 64;
  const int mrow = lane & 15;
  const int kh   = (lane >> 4) << 4;

  v8f acc[2][4] = {};
  const bool fullmask = (causal != 0) && (bn > bm + 127);
  if (!fullmask) {
    if (B_F32) {
      // -------- fp32 weights: cooperative load + convert, reg double buffer
      const int br = tid >> 1;            // 0..127
      const int bc = (tid & 1) << 4;      // 0 or 16
      int brow = bn + br; if (brow > N - 1) brow = N - 1;
      const float* srcrow = (const float*)Bv + (size_t)brow * ldb + bc;
      v8bf c0, c1;
      cvt16(srcrow, c0, c1);
      *(v8bf*)(&Bs[0][br * 32 + bc])     = c0;
      *(v8bf*)(&Bs[0][br * 32 + bc + 8]) = c1;
      int buf = 0;
      for (int k0 = 0; k0 < Kd; k0 += 32) {
        __syncthreads();
        const bool have_next = (k0 + 32) < Kd;
        if (have_next) cvt16(srcrow + k0 + 32, c0, c1);
        v16bf a0 = *(const v16bf*)(A + (size_t)(m0 + mrow) * lda + k0 + kh);
        v16bf a1 = *(const v16bf*)(A + (size_t)(m0 + 16 + mrow) * lda + k0 + kh);
        if (have_next)
          __builtin_prefetch((const void*)(A + (size_t)(m0 + mrow) * lda + k0 + 32 + kh), 0, 3);
#pragma unroll
        for (int j = 0; j < 4; ++j) {
          v16bf b = *(const v16bf*)(&Bs[buf][(wn * 64 + j * 16 + mrow) * 32 + kh]);
          acc[0][j] = wmma_bf16(a0, b, acc[0][j]);
          acc[1][j] = wmma_bf16(a1, b, acc[1][j]);
        }
        if (have_next) {
          *(v8bf*)(&Bs[buf ^ 1][br * 32 + bc])     = c0;
          *(v8bf*)(&Bs[buf ^ 1][br * 32 + bc + 8]) = c1;
        }
        buf ^= 1;
      }
    } else {
      // -------- bf16 B: Tensor Data Mover fills the LDS tile (DMA), zero
      // VALU/VMEM cost; rows >= N are OOB in the descriptor -> zero-filled.
      const bf16* Bb = (const bf16*)Bv;
      const bool issuer = (wave == 0);
      const unsigned lds0 = (unsigned)(unsigned long long)&Bs[0][0];
      const unsigned lds1 = (unsigned)(unsigned long long)&Bs[1][0];
      const unsigned long long gbase =
          (unsigned long long)(const void*)(Bb + (size_t)bn * ldb);
      unsigned rows = (unsigned)(N - bn); if (rows > 128u) rows = 128u;
      if (issuer) tdm_load_tile_bf16(gbase, lds0, rows, (unsigned long long)ldb);
      int buf = 0;
      for (int k0 = 0; k0 < Kd; k0 += 32) {
        if (issuer) __builtin_amdgcn_s_wait_tensorcnt((short)0);
        __syncthreads();
        if ((k0 + 32) < Kd && issuer)
          tdm_load_tile_bf16(gbase + (unsigned long long)(k0 + 32) * 2ull,
                             buf ? lds0 : lds1, rows, (unsigned long long)ldb);
        v16bf a0 = *(const v16bf*)(A + (size_t)(m0 + mrow) * lda + k0 + kh);
        v16bf a1 = *(const v16bf*)(A + (size_t)(m0 + 16 + mrow) * lda + k0 + kh);
#pragma unroll
        for (int j = 0; j < 4; ++j) {
          v16bf b = *(const v16bf*)(&Bs[buf][(wn * 64 + j * 16 + mrow) * 32 + kh]);
          acc[0][j] = wmma_bf16(a0, b, acc[0][j]);
          acc[1][j] = wmma_bf16(a1, b, acc[1][j]);
        }
        buf ^= 1;
      }
    }
  }
  const int moff = (lane >> 4) << 3;     // lanes 16..31 hold rows +8..+15
#pragma unroll
  for (int i2 = 0; i2 < 2; ++i2) {
#pragma unroll
    for (int j = 0; j < 4; ++j) {
      int n = n0 + j * 16 + mrow;
      if (n < N) {
#pragma unroll
        for (int r = 0; r < 8; ++r) {
          int m = m0 + i2 * 16 + moff + r;
          float v = acc[i2][j][r] * alpha;
          if (causal && n > m) v = -1e30f;
          if (row_scale) v *= row_scale[(size_t)m * rs_stride];
          size_t idx = (size_t)m * ldc + n;
          if (accum) C[idx] += v; else C[idx] = v;
        }
      }
    }
  }
}

// ---------------------------------------------------------------------------
// RMS norm over H=2048: one block per row, wave32 shfl reduce + LDS combine.
// ---------------------------------------------------------------------------
__global__ __launch_bounds__(256) void rmsnorm_kernel(
    const float* __restrict__ x, const float* __restrict__ g,
    const float* __restrict__ addend,
    float* __restrict__ outf, bf16* __restrict__ outb)
{
  const int row = blockIdx.x;
  const int tid = threadIdx.x;
  const float* xr = x + (size_t)row * H_;
  float v[8]; float ss = 0.f;
#pragma unroll
  for (int i = 0; i < 8; ++i) { v[i] = xr[tid + i * 256]; ss += v[i] * v[i]; }
#pragma unroll
  for (int off = 16; off > 0; off >>= 1) ss += __shfl_xor(ss, off, 32);
  __shared__ float red[8];
  if ((tid & 31) == 0) red[tid >> 5] = ss;
  __syncthreads();
  ss = 0.f;
#pragma unroll
  for (int i = 0; i < 8; ++i) ss += red[i];
  const float inv = rsqrtf(ss * (1.0f / H_) + EPS_);
#pragma unroll
  for (int i = 0; i < 8; ++i) {
    int c = tid + i * 256;
    float o = v[i] * inv * g[c];
    if (outb) {
      outb[(size_t)row * H_ + c] = (bf16)o;
    } else {
      if (addend) o += addend[(size_t)row * H_ + c];
      outf[(size_t)row * H_ + c] = o;
    }
  }
}

// Row softmax over T=2048 (mask pre-folded as -1e30); emits bf16 probs.
__global__ __launch_bounds__(256) void softmax_kernel(
    const float* __restrict__ S, bf16* __restrict__ P)
{
  const int row = blockIdx.x;
  const int tid = threadIdx.x;
  const float* s = S + (size_t)row * T_;
  bf16* p = P + (size_t)row * T_;
  float v[8]; float mx = -3.4e38f;
#pragma unroll
  for (int i = 0; i < 8; ++i) { v[i] = s[tid + i * 256]; mx = fmaxf(mx, v[i]); }
#pragma unroll
  for (int off = 16; off > 0; off >>= 1) mx = fmaxf(mx, __shfl_xor(mx, off, 32));
  __shared__ float red[8];
  if ((tid & 31) == 0) red[tid >> 5] = mx;
  __syncthreads();
#pragma unroll
  for (int i = 0; i < 8; ++i) mx = fmaxf(mx, red[i]);
  float sum = 0.f;
#pragma unroll
  for (int i = 0; i < 8; ++i) { v[i] = __expf(v[i] - mx); sum += v[i]; }
#pragma unroll
  for (int off = 16; off > 0; off >>= 1) sum += __shfl_xor(sum, off, 32);
  __syncthreads();
  if ((tid & 31) == 0) red[tid >> 5] = sum;
  __syncthreads();
  sum = 0.f;
#pragma unroll
  for (int i = 0; i < 8; ++i) sum += red[i];
  const float inv = 1.0f / sum;
#pragma unroll
  for (int i = 0; i < 8; ++i) p[tid + i * 256] = (bf16)(v[i] * inv);
}

// RoPE on q and k straight out of the fused qkv buffer -> bf16.
__global__ __launch_bounds__(256) void rope_kernel(
    const float* __restrict__ qkv, const int* __restrict__ pos,
    bf16* __restrict__ qo, bf16* __restrict__ ko)
{
  int idx = blockIdx.x * 256 + threadIdx.x;      // T*(NH+NKV)*32
  int d  = idx & 31;
  int hh = (idx >> 5) % (NH_ + NKV_);
  int t  = idx / (32 * (NH_ + NKV_));
  const float* base; bf16* out;
  if (hh < NH_) {
    base = qkv + (size_t)t * QKVN_ + hh * HD_;
    out  = qo  + (size_t)t * (NH_ * HD_) + hh * HD_;
  } else {
    int kv = hh - NH_;
    base = qkv + (size_t)t * QKVN_ + NH_ * HD_ + kv * HD_;
    out  = ko  + (size_t)t * (NKV_ * HD_) + kv * HD_;
  }
  float inv_freq = __powf(10000.0f, -(float)d * (1.0f / 32.0f));
  float fr = (float)pos[t] * inv_freq;
  float sn, cs; __sincosf(fr, &sn, &cs);
  float x1 = base[d], x2 = base[d + 32];
  out[d]      = (bf16)(x1 * cs - x2 * sn);
  out[d + 32] = (bf16)(x2 * cs + x1 * sn);
}

// V transpose into [NKV][HD][T] bf16 so PV is also a TN WMMA GEMM.
__global__ __launch_bounds__(256) void vt_kernel(
    const float* __restrict__ qkv, bf16* __restrict__ vt)
{
  int idx = blockIdx.x * 256 + threadIdx.x;      // NKV*HD*T
  int t    = idx & (T_ - 1);
  int rest = idx >> 11;
  int d    = rest & (HD_ - 1);
  int kv   = rest >> 6;
  vt[((size_t)kv * HD_ + d) * T_ + t] =
      (bf16)qkv[(size_t)t * QKVN_ + (NH_ + NKV_) * HD_ + kv * HD_ + d];
}

// Router logits: tiny (T x 8) — plain dot products.
__global__ __launch_bounds__(256) void gate_kernel(
    const bf16* __restrict__ hmoe, const float* __restrict__ gw,
    float* __restrict__ logits)
{
  int idx = blockIdx.x * 256 + threadIdx.x;      // T*E
  int e = idx & (E_ - 1);
  int t = idx >> 3;
  const bf16* hr = hmoe + (size_t)t * H_;
  const float* wr = gw + (size_t)e * H_;
  float s = 0.f;
  for (int j = 0; j < H_; ++j) s += (float)hr[j] * wr[j];
  logits[(size_t)t * E_ + e] = s;
}

// Top-2 + softmax over the two winners -> dense per-(token,expert) weights.
__global__ __launch_bounds__(256) void top2_kernel(
    const float* __restrict__ logits, float* __restrict__ wexp)
{
  int t = blockIdx.x * 256 + threadIdx.x;
  if (t >= T_) return;
  const float* l = logits + (size_t)t * E_;
  int i1 = 0; float m1 = l[0];
  for (int e = 1; e < E_; ++e) if (l[e] > m1) { m1 = l[e]; i1 = e; }
  int i2 = -1; float m2 = -3.4e38f;
  for (int e = 0; e < E_; ++e) { if (e == i1) continue; if (l[e] > m2) { m2 = l[e]; i2 = e; } }
  float e2 = __expf(m2 - m1);
  float w1 = 1.0f / (1.0f + e2);
  float w2 = e2 * w1;
  for (int e = 0; e < E_; ++e)
    wexp[(size_t)t * E_ + e] = (e == i1) ? w1 : (e == i2) ? w2 : 0.0f;
}

// y = silu(gate) * up, emitted as bf16 for the down-proj WMMA GEMM.
__global__ __launch_bounds__(256) void silu_mul_kernel(
    const float* __restrict__ gu, bf16* __restrict__ y)
{
  int idx = blockIdx.x * 256 + threadIdx.x;      // T*I
  int t = idx >> 12;
  int j = idx & (I_ - 1);
  float g = gu[(size_t)t * (2 * I_) + j];
  float u = gu[(size_t)t * (2 * I_) + I_ + j];
  float s = g / (1.0f + __expf(-g));
  y[idx] = (bf16)(s * u);
}

__global__ __launch_bounds__(256) void cvt_bf16_kernel(
    const float* __restrict__ in, bf16* __restrict__ out, int n)
{
  int i = blockIdx.x * 256 + threadIdx.x;
  if (i < n) out[i] = (bf16)in[i];
}

__global__ __launch_bounds__(256) void add_inplace_kernel(
    float* __restrict__ x, const float* __restrict__ y, int n)
{
  int i = blockIdx.x * 256 + threadIdx.x;
  if (i < n) x[i] += y[i];
}

// ---------------------------------------------------------------------------
extern "C" void kernel_launch(void* const* d_in, const int* in_sizes, int n_in,
                              void* d_out, int out_size, void* d_ws, size_t ws_size,
                              hipStream_t stream)
{
  (void)in_sizes; (void)n_in; (void)out_size; (void)ws_size;
  const int*   positions   = (const int*)d_in[0];
  const float* hidden      = (const float*)d_in[1];
  const float* qkv_w       = (const float*)d_in[2];
  const float* o_w         = (const float*)d_in[3];
  const float* gate_w      = (const float*)d_in[4];
  const float* ws_w        = (const float*)d_in[5];   // [E][2I][H]
  const float* w2s_w       = (const float*)d_in[6];   // [E][H][I]
  const float* pre_attn_g  = (const float*)d_in[7];
  const float* post_attn_g = (const float*)d_in[8];
  const float* pre_moe_g   = (const float*)d_in[9];
  const float* post_moe_g  = (const float*)d_in[10];

  float* out_h = (float*)d_out;                       // h
  float* out_r = out_h + (size_t)T_ * H_;             // residual = rmsnorm(x)

  char* wsbase = (char*)d_ws;
  size_t off = 0;
  auto alloc = [&](size_t bytes) -> char* {
    char* p = wsbase + off;
    off += (bytes + 255) & ~(size_t)255;
    return p;
  };
  bf16*  hn     = (bf16*) alloc((size_t)T_ * H_ * 2);
  float* qkvb   = (float*)alloc((size_t)T_ * QKVN_ * 4);  // reused as scores
  bf16*  qb     = (bf16*) alloc((size_t)T_ * NH_ * HD_ * 2);
  bf16*  kb     = (bf16*) alloc((size_t)T_ * NKV_ * HD_ * 2);
  bf16*  vtb    = (bf16*) alloc((size_t)NKV_ * HD_ * T_ * 2);
  bf16*  probs  = (bf16*) alloc((size_t)T_ * T_ * 2);
  float* attnf  = (float*)alloc((size_t)T_ * NH_ * HD_ * 4);
  bf16*  attnb  = (bf16*) alloc((size_t)T_ * NH_ * HD_ * 2);
  float* xf     = (float*)alloc((size_t)T_ * H_ * 4);
  bf16*  hmoe   = (bf16*) alloc((size_t)T_ * H_ * 2);
  float* logits = (float*)alloc((size_t)T_ * E_ * 4);
  float* wexp   = (float*)alloc((size_t)T_ * E_ * 4);
  float* gu     = (float*)alloc((size_t)T_ * 2 * I_ * 4);
  bf16*  yb     = (bf16*) alloc((size_t)T_ * I_ * 2);
  float* moeo   = (float*)alloc((size_t)T_ * H_ * 4);
  float* scores = qkvb;   // qkv buffer is dead after rope + v-transpose

  dim3 blk(256);

  // 1) pre-attention RMS norm -> bf16
  rmsnorm_kernel<<<T_, blk, 0, stream>>>(hidden, pre_attn_g, nullptr, nullptr, hn);
  // 2) fused QKV projection
  wmma_gemm_tn<true><<<dim3(T_ / 128, QKVN_ / 128), blk, 0, stream>>>(
      hn, H_, qkv_w, H_, qkvb, QKVN_, T_, QKVN_, H_, 1.0f, 0, nullptr, 0, 0);
  // 3) RoPE -> bf16 q,k ; 4) V transpose -> bf16 [NKV][HD][T]
  rope_kernel<<<(T_ * (NH_ + NKV_) * 32) / 256, blk, 0, stream>>>(qkvb, positions, qb, kb);
  vt_kernel<<<(NKV_ * HD_ * T_) / 256, blk, 0, stream>>>(qkvb, vtb);

  // 5) attention, one head at a time (score buffer reused; stream-ordered)
  const float scale = 0.125f;   // HD^-0.5
  for (int h = 0; h < NH_; ++h) {
    int kv = h / (NH_ / NKV_);
    wmma_gemm_tn<false><<<dim3(T_ / 128, T_ / 128), blk, 0, stream>>>(
        qb + h * HD_, NH_ * HD_, kb + kv * HD_, NKV_ * HD_,
        scores, T_, T_, T_, HD_, scale, 1, nullptr, 0, 0);
    softmax_kernel<<<T_, blk, 0, stream>>>(scores, probs);
    wmma_gemm_tn<false><<<dim3(T_ / 128, 1), blk, 0, stream>>>(
        probs, T_, vtb + (size_t)kv * HD_ * T_, T_,
        attnf + h * HD_, NH_ * HD_, T_, HD_, T_, 1.0f, 0, nullptr, 0, 0);
  }
  cvt_bf16_kernel<<<(T_ * NH_ * HD_) / 256, blk, 0, stream>>>(attnf, attnb, T_ * NH_ * HD_);

  // 6) output projection + residual add
  wmma_gemm_tn<true><<<dim3(T_ / 128, H_ / 128), blk, 0, stream>>>(
      attnb, NH_ * HD_, o_w, NH_ * HD_, xf, H_, T_, H_, NH_ * HD_, 1.0f, 0, nullptr, 0, 0);
  add_inplace_kernel<<<(T_ * H_) / 256, blk, 0, stream>>>(xf, hidden, T_ * H_);

  // 7) post-attention RMS norm -> residual output (fp32)
  rmsnorm_kernel<<<T_, blk, 0, stream>>>(xf, post_attn_g, nullptr, out_r, nullptr);
  // 8) pre-MoE RMS norm -> bf16
  rmsnorm_kernel<<<T_, blk, 0, stream>>>(out_r, pre_moe_g, nullptr, nullptr, hmoe);

  // 9) router
  gate_kernel<<<(T_ * E_) / 256, blk, 0, stream>>>(hmoe, gate_w, logits);
  top2_kernel<<<T_ / 256, blk, 0, stream>>>(logits, wexp);

  // 10) experts: up/gate GEMM -> silu*up -> down GEMM with fused routing
  //     weight (row_scale) and accumulation across experts.
  for (int e = 0; e < E_; ++e) {
    wmma_gemm_tn<true><<<dim3(T_ / 128, (2 * I_) / 128), blk, 0, stream>>>(
        hmoe, H_, ws_w + (size_t)e * 2 * I_ * H_, H_,
        gu, 2 * I_, T_, 2 * I_, H_, 1.0f, 0, nullptr, 0, 0);
    silu_mul_kernel<<<(T_ * I_) / 256, blk, 0, stream>>>(gu, yb);
    wmma_gemm_tn<true><<<dim3(T_ / 128, H_ / 128), blk, 0, stream>>>(
        yb, I_, w2s_w + (size_t)e * H_ * I_, I_,
        moeo, H_, T_, H_, I_, 1.0f, 0, wexp + e, E_, e == 0 ? 0 : 1);
  }

  // 11) h = residual + rmsnorm(moe_out, post_moe_g)
  rmsnorm_kernel<<<T_, blk, 0, stream>>>(moeo, post_moe_g, out_r, out_h, nullptr);
}
